// MMFConv2d_25305947308550
// MI455X (gfx1250) — matrix operand order repaired and checked by reading, not compiled, not run on hardware
//
#include <hip/hip_runtime.h>

typedef __attribute__((ext_vector_type(16))) _Float16 v16h;
typedef __attribute__((ext_vector_type(8)))  _Float16 v8h;
typedef __attribute__((ext_vector_type(8)))  float    v8f;

#define CO    256
#define CIN   128
#define HW    56
#define NPIX  (56 * 56)
#define KDIM  128
#define LDS_STRIDE 136   // 128 halves + pad; 272 B = 17*16 B -> 16B-aligned columns

// ---------------------------------------------------------------------------
// Kernel 1: W_eff[co][ci] = sum over 3x3 of ternary weights, stored as f16.
// Integers |v|<=9 are exact in f16.
// ---------------------------------------------------------------------------
__global__ void weff_fold_kernel(const float* __restrict__ w,
                                 _Float16* __restrict__ wh) {
    int idx = blockIdx.x * blockDim.x + threadIdx.x;   // co*128 + ci
    if (idx >= CO * CIN) return;
    const float* p = w + (size_t)idx * 9;
    float s = 0.0f;
#pragma unroll
    for (int i = 0; i < 9; ++i) s += p[i];
    wh[idx] = (_Float16)s;
}

// ---------------------------------------------------------------------------
// Kernel 2: per-block GEMM tile. Block = (batch b, 16 output pixels n0..n0+15).
// 8 waves; wave w computes output channels [32w, 32w+32) via two 16x16 WMMA
// accumulators, K = 128 in four K32 steps of v_wmma_f32_16x16x32_f16.
// ---------------------------------------------------------------------------
__global__ __launch_bounds__(256)
void mmf_conv_wmma_kernel(const float* __restrict__ x,
                          const _Float16* __restrict__ wh,
                          const float* __restrict__ bias,
                          const float* __restrict__ scale,
                          float* __restrict__ out) {
    __shared__ __align__(16) _Float16 ub[16 * LDS_STRIDE];

    const int b  = blockIdx.y;
    const int n0 = blockIdx.x * 16;
    const int t  = threadIdx.x;
    const float sc = scale[0];

    // ---- Stage B tile into LDS: U[r][j], r=0..127 (im2col row), j=0..15 (pixel).
    // Stored column-major: ub[j*LDS_STRIDE + r], so a B-fragment read is two runs of 8
    // contiguous K halves down one column.
#pragma unroll
    for (int i = 0; i < 8; ++i) {
        int e = t + 256 * i;          // 0..2047
        int r = e & 127;
        int j = e >> 7;
        int n  = n0 + j;
        int ho = n / HW, wo = n % HW;
        int c  = r / 9,  p  = r % 9;
        int ih = ho + p / 3 - 1;      // pad = 1
        int iw = wo + p % 3 - 1;
        float v = 0.0f;
        if ((unsigned)ih < (unsigned)HW && (unsigned)iw < (unsigned)HW)
            v = x[(((size_t)b * CIN + c) * HW + ih) * HW + iw];
        ub[j * LDS_STRIDE + r] = (_Float16)v;
    }
    __syncthreads();

    const int wave  = t >> 5;
    const int lane  = t & 31;
    const int lhalf = lane >> 4;      // 0: lanes 0-15, 1: lanes 16-31
    const int lrow  = lane & 15;

    // B fragment source: column n = lrow, K base = lhalf*8 (same K order as A:
    // lanes 0-15 hold K {0..7, 16..23}, lanes 16-31 hold K {8..15, 24..31}).
    const _Float16* bcol = &ub[lrow * LDS_STRIDE + lhalf * 8];

    // A fragment source: row m = co. ISA A layout for 16-bit 16x32: lanes 0-15
    // hold K {0..7, 16..23}, lanes 16-31 hold K {8..15, 24..31}.
    const int co0 = wave * 32 + lrow;
    const int co1 = co0 + 16;
    const _Float16* arow0 = wh + (size_t)co0 * KDIM + lhalf * 8;
    const _Float16* arow1 = wh + (size_t)co1 * KDIM + lhalf * 8;

    v8f acc0 = {};
    v8f acc1 = {};
#pragma unroll
    for (int kc = 0; kc < 4; ++kc) {
        v8h a0lo = *(const v8h*)(arow0 + kc * 32);
        v8h a0hi = *(const v8h*)(arow0 + kc * 32 + 16);
        v8h a1lo = *(const v8h*)(arow1 + kc * 32);
        v8h a1hi = *(const v8h*)(arow1 + kc * 32 + 16);
        v16h afrag0 = __builtin_shufflevector(a0lo, a0hi,
                        0,1,2,3,4,5,6,7,8,9,10,11,12,13,14,15);
        v16h afrag1 = __builtin_shufflevector(a1lo, a1hi,
                        0,1,2,3,4,5,6,7,8,9,10,11,12,13,14,15);

        v8h blo = *(const v8h*)(bcol + kc * 32);
        v8h bhi = *(const v8h*)(bcol + kc * 32 + 16);
        v16h bfrag = __builtin_shufflevector(blo, bhi,
                        0,1,2,3,4,5,6,7,8,9,10,11,12,13,14,15);

        acc0 = __builtin_amdgcn_wmma_f32_16x16x32_f16(
                   false, afrag0, false, bfrag, (short)0, acc0, false, false);
        acc1 = __builtin_amdgcn_wmma_f32_16x16x32_f16(
                   false, afrag1, false, bfrag, (short)0, acc1, false, false);
    }

    // ---- Store D. C/D layout: VGPR v, lanes 0-15 -> (M=v, N=lane);
    // lanes 16-31 -> (M=v+8, N=lane-16).
    const size_t outbase = (size_t)b * CO * NPIX + (size_t)(n0 + lrow);
#pragma unroll
    for (int v = 0; v < 8; ++v) {
        int m0 = wave * 32 + v + lhalf * 8;
        int m1 = m0 + 16;
        out[outbase + (size_t)m0 * NPIX] = sc * acc0[v] + bias[m0];
        out[outbase + (size_t)m1 * NPIX] = sc * acc1[v] + bias[m1];
    }
}

// ---------------------------------------------------------------------------
extern "C" void kernel_launch(void* const* d_in, const int* in_sizes, int n_in,
                              void* d_out, int out_size, void* d_ws, size_t ws_size,
                              hipStream_t stream) {
    const float* x      = (const float*)d_in[0];   // (32,128,56,56)
    const float* weight = (const float*)d_in[1];   // (256,128,3,3)
    const float* bias   = (const float*)d_in[2];   // (256,)
    const float* scale  = (const float*)d_in[3];   // scalar
    float*       out    = (float*)d_out;           // (32,256,56,56)

    _Float16* wh = (_Float16*)d_ws;                // 256*128 f16 = 64 KB

    // Fold ternary 3x3 weights -> f16 W_eff (exact).
    weff_fold_kernel<<<(CO * CIN + 255) / 256, 256, 0, stream>>>(weight, wh);

    // GEMM: 196 pixel tiles per batch, 32 batches; 256 threads per block.
    dim3 grid(NPIX / 16, 32);
    mmf_conv_wmma_kernel<<<grid, 256, 0, stream>>>(x, wh, bias, scale, out);
}
